// GraLstmEnc_7962869367596
// MI455X (gfx1250) — matrix operand order repaired and checked by reading, not compiled, not run on hardware
//
#include <hip/hip_runtime.h>
#include <hip/hip_bf16.h>

// ---------------- constants ----------------
#define NN   16384      // total nodes
#define EE   262144     // edges
#define SS   256        // graphs (batch)
#define TT   32         // timesteps
#define CENC 128
#define CGNN 128
#define HID  512
#define FC1N 256
#define NC   4
#define DECAY   0.2f
#define THRESH  0.5f

typedef __attribute__((ext_vector_type(16))) _Float16 v16h;
typedef __attribute__((ext_vector_type(8)))  float    v8f;

union HV { v16h v; unsigned u[8]; _Float16 h[16]; };

// ---------------- CDNA5 async copy helpers ----------------
// GLOBAL_LOAD_ASYNC_TO_LDS_B128: memory -> LDS, no VGPR data, tracked by ASYNCcnt.
// %0 = per-lane LDS byte offset VGPR, %1 = per-lane 64-bit global address.
__device__ inline void async_ld_b128(unsigned lds_off, const void* gaddr) {
  asm volatile("global_load_async_to_lds_b128 %0, %1, off"
               :: "v"(lds_off), "v"(gaddr) : "memory");
}
__device__ inline void wait_async0() {
  asm volatile("s_wait_asynccnt 0x0" ::: "memory");
}
// generic pointer to LDS keeps the DS byte offset in its low 32 bits
__device__ inline unsigned lds_off32(const void* p) {
  return (unsigned)(size_t)p;
}

// ------------- WMMA operand loaders (layouts per CDNA5 ISA 7.12.2) -------------
// A: 16x32 f16 tile, row-major in LDS with row stride `lda` halfs.
// lane m = lane&15 holds row M=m; lane group g=lane>>4 selects K-halves:
//   u[0..3] -> K = {0,1},{2,3},{4,5},{6,7} + 8g ; u[4..7] -> K = 16 + {0..7} + 8g
__device__ inline v16h lds_load_A(const _Float16* As, int lda) {
  int lane = threadIdx.x & 31;
  int m = lane & 15, g = lane >> 4;
  const _Float16* row = As + m * lda + 8 * g;
  HV r;
#pragma unroll
  for (int i = 0; i < 4; ++i) {
    r.u[i]     = *(const unsigned*)(row + 2 * i);
    r.u[i + 4] = *(const unsigned*)(row + 16 + 2 * i);
  }
  return r.v;
}

// B: 32x16 f16 tile. Stored n-major (64 rows of N) x 32 contiguous K halfs in LDS.
// lane n = lane&15 selects column N; lane group g: u[0..3] -> K = 0..7 + 8g, u[4..7] -> K = 16 + 0..7 + 8g.
__device__ inline v16h lds_load_B(const _Float16* Bs, int nrow) {
  int lane = threadIdx.x & 31;
  int g = lane >> 4;
  const _Float16* row = Bs + nrow * 32 + 8 * g;
  HV r;
#pragma unroll
  for (int v = 0; v < 4; ++v) {
    r.u[v]     = *(const unsigned*)(row + 2 * v);
    r.u[v + 4] = *(const unsigned*)(row + 16 + 2 * v);
  }
  return r.v;
}

#define WMMA_F16(a, b, c) \
  __builtin_amdgcn_wmma_f32_16x16x32_f16(false, (a), false, (b), (short)0, (c), false, false)

// ------------- generic f16 WMMA GEMM: C[MxN] = A[MxK] * Bt[NxK]^T (+bias) (+C) ----
// block = 128 threads (4 waves), block tile 64(M) x 64(N), wave tile 16 x 64.
// Double-buffered LDS, tiles filled with GLOBAL_LOAD_ASYNC_TO_LDS_B128 so the
// async engine fetches tile i+1 while the waves run WMMA on tile i.
__global__ __launch_bounds__(128)
void k_gemm(const _Float16* __restrict__ A, int lda,
            const _Float16* __restrict__ Bt,         // N x K (transposed weights)
            float* __restrict__ C, int ldc,
            const float* __restrict__ bias,
            int M, int N, int K, int accumulate) {
  __shared__ __align__(16) _Float16 As[2][64 * 32];
  __shared__ __align__(16) _Float16 Bs[2][64 * 32];
  const int tid = threadIdx.x;
  const int wave = tid >> 5, lane = tid & 31;
  const int m0 = blockIdx.y * 64;
  const int n0 = blockIdx.x * 64;
  // number of valid 16-wide N subtiles in this block (uniform)
  const int rem = N - n0;
  const int nsub = (rem >= 64) ? 4 : ((rem + 15) >> 4);

  // per-thread cooperative-fill coordinates: 128 threads x 16 halfs = 64x32 tile
  const int idx = tid * 16;                 // 0..2047
  const int row = idx >> 5, col = idx & 31; // col in {0,16}
  const int nrow = n0 + row;
  const bool bvalid = (nrow < N);
  const _Float16* asrc = A + (size_t)(m0 + row) * lda + col;
  const _Float16* bsrc = Bt + (size_t)nrow * K + col;

  auto issue = [&](int buf, int k0) {
    unsigned ao = lds_off32(&As[buf][idx]);
    async_ld_b128(ao,      asrc + k0);
    async_ld_b128(ao + 16, asrc + k0 + 8);
    __builtin_prefetch(asrc + k0 + 64, 0, 1);     // global_prefetch_b8 (L2 warm)
    if (bvalid) {
      unsigned bo = lds_off32(&Bs[buf][idx]);
      async_ld_b128(bo,      bsrc + k0);
      async_ld_b128(bo + 16, bsrc + k0 + 8);
    }
  };

  v8f acc0 = {}, acc1 = {}, acc2 = {}, acc3 = {};

  const int nk = K >> 5;                    // K is a multiple of 32
  issue(0, 0);                              // prologue: fill buffer 0
  wait_async0();
  __syncthreads();

  if (nsub == 4) {
    // fast path (all LSTM-path GEMMs): straight-line inner loop,
    // all 4 B fragments fetched, then 4 back-to-back WMMAs.
    for (int i = 0; i < nk; ++i) {
      const int cur = i & 1;
      if (i + 1 < nk) issue(cur ^ 1, (i + 1) * 32);   // async fetch next tile
      v16h a  = lds_load_A(&As[cur][wave * 16 * 32], 32);
      v16h b0 = lds_load_B(&Bs[cur][0], 0  + (lane & 15));
      v16h b1 = lds_load_B(&Bs[cur][0], 16 + (lane & 15));
      v16h b2 = lds_load_B(&Bs[cur][0], 32 + (lane & 15));
      v16h b3 = lds_load_B(&Bs[cur][0], 48 + (lane & 15));
      acc0 = WMMA_F16(a, b0, acc0);
      acc1 = WMMA_F16(a, b1, acc1);
      acc2 = WMMA_F16(a, b2, acc2);
      acc3 = WMMA_F16(a, b3, acc3);
      wait_async0();          // next tile resident in LDS
      __syncthreads();        // all waves done reading cur, writes visible
    }
  } else {
    // edge path: partial-N block (only the last block of the N=96 GEMM)
    for (int i = 0; i < nk; ++i) {
      const int cur = i & 1;
      if (i + 1 < nk) issue(cur ^ 1, (i + 1) * 32);
      v16h a = lds_load_A(&As[cur][wave * 16 * 32], 32);
      {
        v16h b = lds_load_B(&Bs[cur][0], 0 + (lane & 15));
        acc0 = WMMA_F16(a, b, acc0);
      }
      if (nsub > 1) {
        v16h b = lds_load_B(&Bs[cur][0], 16 + (lane & 15));
        acc1 = WMMA_F16(a, b, acc1);
      }
      if (nsub > 2) {
        v16h b = lds_load_B(&Bs[cur][0], 32 + (lane & 15));
        acc2 = WMMA_F16(a, b, acc2);
      }
      wait_async0();
      __syncthreads();
    }
  }

  // C/D layout: VGPR r holds M = r + 8*(lane>>4), N = lane&15
  const int g = lane >> 4, n = lane & 15;
  const int mrow = m0 + wave * 16 + g * 8;
  v8f accs[4] = {acc0, acc1, acc2, acc3};
#pragma unroll
  for (int j = 0; j < 4; ++j) {
    int nc = n0 + 16 * j + n;
    if (nc >= N) continue;
    float bi = bias ? bias[nc] : 0.0f;
#pragma unroll
    for (int r = 0; r < 8; ++r) {
      size_t off = (size_t)(mrow + r) * ldc + nc;
      float v = accs[j][r] + bi;
      if (accumulate) v += C[off];
      C[off] = v;
    }
  }
}

// ---------------- elementwise / graph kernels ----------------
__global__ void k_zero(float* p, int n) {
  int g = blockIdx.x * blockDim.x + threadIdx.x;
  if (g < n) p[g] = 0.0f;
}

__global__ void k_copy(float* d, const float* s, int n) {
  int g = blockIdx.x * blockDim.x + threadIdx.x;
  if (g < n) d[g] = s[g];
}

__global__ void k_add2(float* d, const float* a, const float* b, int n) {
  int g = blockIdx.x * blockDim.x + threadIdx.x;
  if (g < n) d[g] = a[g] + b[g];
}

// Wt[n*K + k] = (f16) W[k*N + n]
__global__ void k_transpose_h(const float* __restrict__ W, _Float16* __restrict__ Wt,
                              int K, int N) {
  int g = blockIdx.x * blockDim.x + threadIdx.x;
  if (g >= K * N) return;
  int n = g / K, k = g - n * K;
  Wt[g] = (_Float16)W[(size_t)k * N + n];
}

__global__ void k_deg(const int* __restrict__ ei1, float* deg) {
  int g = blockIdx.x * blockDim.x + threadIdx.x;
  if (g >= EE + NN) return;
  int d = (g < EE) ? ei1[g] : (g - EE);
  atomicAdd(deg + d, 1.0f);
}

__global__ void k_dinv(const float* deg, float* dinv) {
  int g = blockIdx.x * blockDim.x + threadIdx.x;
  if (g >= NN) return;
  float d = deg[g];
  dinv[g] = (d > 0.0f) ? rsqrtf(fmaxf(d, 1e-12f)) : 0.0f;
}

// encoding LIF: mem = mem*DECAY*(1-spike) + x_t @ enc_W + enc_b ; spike = mem > 0.5
__global__ void k_enc(const float* __restrict__ x, const float* __restrict__ encW,
                      const float* __restrict__ encb, float* __restrict__ enc_mem,
                      _Float16* __restrict__ enc_spk, int t) {
  int g = blockIdx.x * blockDim.x + threadIdx.x;
  if (g >= NN * CENC) return;
  int n = g >> 7, c = g & 127;
  float acc = encb[c];
#pragma unroll
  for (int ci = 0; ci < 8; ++ci)
    acc = fmaf(x[(size_t)(n * 8 + ci) * TT + t], encW[ci * CENC + c], acc);
  float prev = (float)enc_spk[g];
  float mem = enc_mem[g] * DECAY * (1.0f - prev) + acc;
  enc_mem[g] = mem;
  enc_spk[g] = (_Float16)((mem > THRESH) ? 1.0f : 0.0f);
}

// symnorm scatter: agg[d] += dinv[s]*dinv[d] * bases[s]  (bases = bc[:, 0:64])
__global__ void k_scatter(const int* __restrict__ ei0, const int* __restrict__ ei1,
                          const float* __restrict__ dinv, const float* __restrict__ bc,
                          float* __restrict__ agg) {
  int g = blockIdx.x * blockDim.x + threadIdx.x;
  int e = g >> 5, l = g & 31;
  if (e >= EE + NN) return;
  int s, d;
  if (e < EE) { s = ei0[e]; d = ei1[e]; } else { s = e - EE; d = s; }
  float w = dinv[s] * dinv[d];
  const float* b = bc + (size_t)s * 96;
  atomicAdd(agg + (size_t)d * 64 + l,      w * b[l]);
  atomicAdd(agg + (size_t)d * 64 + l + 32, w * b[l + 32]);
}

// conv[n,h,f] = sum_b comb[n,h,b]*agg[n,b,f] + bias; then LIF -> c1_spike (= xl, f16)
__global__ void k_conv_c1(const float* __restrict__ bc, const float* __restrict__ agg,
                          const float* __restrict__ convb, float* __restrict__ c1_mem,
                          _Float16* __restrict__ c1_spk) {
  int g = blockIdx.x * blockDim.x + threadIdx.x;
  if (g >= NN * CGNN) return;
  int n = g >> 7, c = g & 127, h = c >> 4, f = c & 15;
  const float* combn = bc + (size_t)n * 96 + 64;
  const float* aggn  = agg + (size_t)n * 64;
  float acc = convb[c];
#pragma unroll
  for (int b = 0; b < 4; ++b)
    acc = fmaf(combn[h * 4 + b], aggn[b * 16 + f], acc);
  float prev = (float)c1_spk[g];
  float mem = c1_mem[g] * DECAY * (1.0f - prev) + acc;
  c1_mem[g] = mem;
  c1_spk[g] = (_Float16)((mem > THRESH) ? 1.0f : 0.0f);
}

// spiking LSTM update: i,f,g,o = (gate>0); lc = f*lc + i*g; lh = lc*o (exact in f16)
__global__ void k_gate(const float* __restrict__ gates, float* __restrict__ lc,
                       _Float16* __restrict__ lh) {
  int g = blockIdx.x * blockDim.x + threadIdx.x;
  if (g >= SS * HID) return;
  int s = g >> 9, j = g & 511;
  const float* gr = gates + (size_t)s * (4 * HID);
  float gi = (gr[j]            > 0.0f) ? 1.0f : 0.0f;
  float gf = (gr[HID + j]      > 0.0f) ? 1.0f : 0.0f;
  float gg = (gr[2 * HID + j]  > 0.0f) ? 1.0f : 0.0f;
  float go = (gr[3 * HID + j]  > 0.0f) ? 1.0f : 0.0f;
  float c = gf * lc[g] + gi * gg;
  lc[g] = c;
  lh[g] = (_Float16)(c * go);
}

__global__ void k_h1(const float* __restrict__ fc1o, float* __restrict__ h1_mem,
                     _Float16* __restrict__ h1_spk) {
  int g = blockIdx.x * blockDim.x + threadIdx.x;
  if (g >= SS * FC1N) return;
  float prev = (float)h1_spk[g];
  float mem = h1_mem[g] * DECAY * (1.0f - prev) + fc1o[g];
  h1_mem[g] = mem;
  h1_spk[g] = (_Float16)((mem > THRESH) ? 1.0f : 0.0f);
}

__global__ void k_h2(const _Float16* __restrict__ h1_spk, const float* __restrict__ fc2W,
                     const float* __restrict__ fc2b, float* __restrict__ h2_mem,
                     float* __restrict__ h2_spk, float* __restrict__ h2_sum) {
  int g = blockIdx.x * blockDim.x + threadIdx.x;
  if (g >= SS * NC) return;
  int s = g >> 2, c = g & 3;
  const _Float16* hs = h1_spk + (size_t)s * FC1N;
  float acc = fc2b[c];
  for (int k = 0; k < FC1N; ++k)
    acc = fmaf((float)hs[k], fc2W[k * NC + c], acc);
  float prev = h2_spk[g];
  float mem = h2_mem[g] * DECAY * (1.0f - prev) + acc;
  h2_mem[g] = mem;
  float sp = (mem > THRESH) ? 1.0f : 0.0f;
  h2_spk[g] = sp;
  h2_sum[g] += sp;
}

__global__ void k_final(const float* h2_sum, float* out) {
  int g = blockIdx.x * blockDim.x + threadIdx.x;
  if (g < SS * NC) out[g] = h2_sum[g] * (1.0f / (float)TT);
}

// ---------------- host launcher ----------------
extern "C" void kernel_launch(void* const* d_in, const int* in_sizes, int n_in,
                              void* d_out, int out_size, void* d_ws, size_t ws_size,
                              hipStream_t stream) {
  (void)in_sizes; (void)n_in; (void)out_size; (void)ws_size;
  const float* x      = (const float*)d_in[0];
  const int*   ei     = (const int*)  d_in[1];
  const float* encW   = (const float*)d_in[2];
  const float* encb   = (const float*)d_in[3];
  const float* basesW = (const float*)d_in[4];
  const float* basesb = (const float*)d_in[5];
  const float* combW  = (const float*)d_in[6];
  const float* combb  = (const float*)d_in[7];
  const float* convb  = (const float*)d_in[8];
  const float* Wih    = (const float*)d_in[9];
  const float* Whh    = (const float*)d_in[10];
  const float* bih    = (const float*)d_in[11];
  const float* bhh    = (const float*)d_in[12];
  const float* fc1W   = (const float*)d_in[13];
  const float* fc1b   = (const float*)d_in[14];
  const float* fc2W   = (const float*)d_in[15];
  const float* fc2b   = (const float*)d_in[16];
  float* out = (float*)d_out;

  // workspace carve-out
  char* base = (char*)d_ws; size_t off = 0;
  auto nxt = [&](size_t bytes) -> void* {
    void* p = base + off;
    off += (bytes + 255) & ~(size_t)255;
    return p;
  };
  float*    deg     = (float*)   nxt((size_t)NN * 4);
  float*    dinv    = (float*)   nxt((size_t)NN * 4);
  float*    enc_mem = (float*)   nxt((size_t)NN * CENC * 4);
  _Float16* enc_spk = (_Float16*)nxt((size_t)NN * CENC * 2);
  float*    bc      = (float*)   nxt((size_t)NN * 96 * 4);   // [bases(64)|comb(32)]
  float*    agg     = (float*)   nxt((size_t)NN * 64 * 4);
  float*    c1_mem  = (float*)   nxt((size_t)NN * CGNN * 4);
  _Float16* xl      = (_Float16*)nxt((size_t)NN * CGNN * 2); // c1 spikes == LSTM input
  float*    gates   = (float*)   nxt((size_t)SS * 4 * HID * 4);
  float*    lc      = (float*)   nxt((size_t)SS * HID * 4);
  _Float16* lh      = (_Float16*)nxt((size_t)SS * HID * 2);
  float*    fc1o    = (float*)   nxt((size_t)SS * FC1N * 4);
  float*    h1_mem  = (float*)   nxt((size_t)SS * FC1N * 4);
  _Float16* h1_spk  = (_Float16*)nxt((size_t)SS * FC1N * 2);
  float*    h2_mem  = (float*)   nxt((size_t)SS * NC * 4);
  float*    h2_spk  = (float*)   nxt((size_t)SS * NC * 4);
  float*    h2_sum  = (float*)   nxt((size_t)SS * NC * 4);
  _Float16* WbcT    = (_Float16*)nxt((size_t)96 * CENC * 2);
  float*    bcb     = (float*)   nxt((size_t)96 * 4);
  _Float16* WihT    = (_Float16*)nxt((size_t)2048 * 8192 * 2);
  _Float16* WhhT    = (_Float16*)nxt((size_t)2048 * 512 * 2);
  _Float16* fc1WT   = (_Float16*)nxt((size_t)256 * 512 * 2);
  float*    biasG   = (float*)   nxt((size_t)2048 * 4);

  auto blocks = [](long n) { return dim3((unsigned)((n + 255) / 256)); };
  auto zero = [&](void* p, long f32n) {
    k_zero<<<blocks(f32n), 256, 0, stream>>>((float*)p, (int)f32n);
  };

  // ---- one-time init (re-done every call: deterministic) ----
  zero(deg, NN);
  zero(enc_mem, (long)NN * CENC); zero(enc_spk, (long)NN * CENC / 2);
  zero(c1_mem, (long)NN * CGNN);  zero(xl, (long)NN * CGNN / 2);
  zero(lc, (long)SS * HID);       zero(lh, (long)SS * HID / 2);
  zero(h1_mem, (long)SS * FC1N);  zero(h1_spk, (long)SS * FC1N / 2);
  zero(h2_mem, SS * NC); zero(h2_spk, SS * NC); zero(h2_sum, SS * NC);

  k_deg <<<blocks(EE + NN), 256, 0, stream>>>(ei + EE, deg);
  k_dinv<<<blocks(NN), 256, 0, stream>>>(deg, dinv);

  k_transpose_h<<<blocks(128L * 64),    256, 0, stream>>>(basesW, WbcT,             CENC, 64);
  k_transpose_h<<<blocks(128L * 32),    256, 0, stream>>>(combW,  WbcT + 64 * CENC, CENC, 32);
  k_transpose_h<<<blocks(8192L * 2048), 256, 0, stream>>>(Wih,    WihT,  8192, 2048);
  k_transpose_h<<<blocks(512L * 2048),  256, 0, stream>>>(Whh,    WhhT,  512,  2048);
  k_transpose_h<<<blocks(512L * 256),   256, 0, stream>>>(fc1W,   fc1WT, 512,  256);
  k_copy<<<blocks(64), 256, 0, stream>>>(bcb,      basesb, 64);
  k_copy<<<blocks(32), 256, 0, stream>>>(bcb + 64, combb,  32);
  k_add2<<<blocks(2048), 256, 0, stream>>>(biasG, bih, bhh, 2048);

  auto gemm = [&](const _Float16* A, int K, const _Float16* Bt, float* C,
                  const float* bias, int M, int Ncols, int accum) {
    dim3 grid((Ncols + 63) / 64, M / 64);
    k_gemm<<<grid, 128, 0, stream>>>(A, K, Bt, C, Ncols, bias, M, Ncols, K, accum);
  };

  // ---- recurrent timeline ----
  for (int t = 0; t < TT; ++t) {
    k_enc<<<blocks((long)NN * CENC), 256, 0, stream>>>(x, encW, encb, enc_mem, enc_spk, t);
    zero(agg, (long)NN * 64);
    // [bases|comb] = enc_spike @ [bases_W|comb_W]    (WMMA, K=128, N=96)
    gemm(enc_spk, CENC, WbcT, bc, bcb, NN, 96, 0);
    k_scatter<<<blocks((long)(EE + NN) * 32), 256, 0, stream>>>(ei, ei + EE, dinv, bc, agg);
    k_conv_c1<<<blocks((long)NN * CGNN), 256, 0, stream>>>(bc, agg, convb, c1_mem, xl);
    // gates = xl @ W_ih + (b_ih + b_hh)              (WMMA, dominant: 256x2048x8192)
    gemm(xl, 8192, WihT, gates, biasG, SS, 2048, 0);
    // gates += lh @ W_hh                             (WMMA, accumulate)
    gemm(lh, HID, WhhT, gates, nullptr, SS, 2048, 1);
    k_gate<<<blocks((long)SS * HID), 256, 0, stream>>>(gates, lc, lh);
    // fc1 = lh @ fc1_W + fc1_b                       (WMMA)
    gemm(lh, HID, fc1WT, fc1o, fc1b, SS, FC1N, 0);
    k_h1<<<blocks((long)SS * FC1N), 256, 0, stream>>>(fc1o, h1_mem, h1_spk);
    k_h2<<<blocks(SS * NC), 256, 0, stream>>>(h1_spk, fc2W, fc2b, h2_mem, h2_spk, h2_sum);
  }
  k_final<<<blocks(SS * NC), 256, 0, stream>>>(h2_sum, out);
}
